// MultiHeadAttention_27307402068219
// MI455X (gfx1250) — matrix operand order repaired and checked by reading, not compiled, not run on hardware
//
#include <hip/hip_runtime.h>

typedef __bf16 v16bf __attribute__((ext_vector_type(16)));
typedef __bf16 v8bf  __attribute__((ext_vector_type(8)));
typedef float  v8f   __attribute__((ext_vector_type(8)));

constexpr int BATCH  = 4;
constexpr int SEQ    = 2048;
constexpr int DIM    = 1024;
constexpr int NHEAD  = 16;
constexpr int HDIM   = 64;
constexpr int ROWS   = BATCH * SEQ;   // 8192
constexpr int KT     = DIM / 32;      // 32  k-tiles per GEMM
constexpr int NT     = DIM / 16;      // 64  n-tiles per GEMM row
constexpr int QTILES = SEQ / 16;      // 128
// register blocking for the dense GEMMs: 2 row-tiles x 4 col-tiles per wave
constexpr int MBLK   = 2;
constexpr int NBLK   = 4;
constexpr int RTB    = ROWS / (16 * MBLK);  // 256
constexpr int NTB    = DIM / (16 * NBLK);   // 16

#define WMMA_BF16(a, b, c) \
  __builtin_amdgcn_wmma_f32_16x16x32_bf16(false, (a), false, (b), (short)0, (c), false, false)

// k-offset pattern of the 16-bit 16x32 A fragment (contiguous, matches B packing):
// k = 16*h + e   (h = lane>>4)
__device__ __forceinline__ int koff(int e, int h) {
  return 16 * h + e;
}

// ---------------- pack: row-major [rows, kdim] -> A-fragment order ----------------
template <typename TIN>
__global__ __launch_bounds__(256) void pack_a_kernel(const TIN* __restrict__ X,
                                                     __bf16* __restrict__ out,
                                                     int rows, int kdim) {
  int tid = blockIdx.x * blockDim.x + threadIdx.x;
  int ktc = kdim / 32;
  int total = (rows / 16) * ktc * 32;
  if (tid >= total) return;
  int lane = tid & 31;
  int idx  = tid >> 5;
  int kt = idx % ktc;
  int rt = idx / ktc;
  int h = lane >> 4;
  int r = rt * 16 + (lane & 15);
  __bf16* o = out + (size_t)tid * 16;
#pragma unroll
  for (int e = 0; e < 16; ++e)
    o[e] = (__bf16)(float)X[(size_t)r * kdim + kt * 32 + koff(e, h)];
}

// ---------------- pack: row-major W [kdim, ndim] -> B-fragment order ----------------
__global__ __launch_bounds__(256) void pack_b_kernel(const float* __restrict__ W,
                                                     __bf16* __restrict__ out,
                                                     int kdim, int ndim) {
  int tid = blockIdx.x * blockDim.x + threadIdx.x;
  int ntc = ndim / 16;
  int total = (kdim / 32) * ntc * 32;
  if (tid >= total) return;
  int lane = tid & 31;
  int idx  = tid >> 5;
  int nt = idx % ntc;
  int kt = idx / ntc;
  int h = lane >> 4;
  int n = nt * 16 + (lane & 15);
  __bf16* o = out + (size_t)tid * 16;
#pragma unroll
  for (int e = 0; e < 16; ++e)
    o[e] = (__bf16)W[(size_t)(kt * 32 + 16 * h + e) * ndim + n];
}

// ------------- projection GEMM (2x4 register-blocked):
//   Apk[ROWS,DIM] x Bpk[DIM,DIM] -> bf16 head layout
// vt_layout==0 : out[b,h,s,hd]   (Q and K)
// vt_layout==1 : out[b,h,hd,s]   (V transposed: B-operand friendly for P*V)
__global__ __launch_bounds__(256) void gemm_qkv_kernel(const __bf16* __restrict__ Apk,
                                                       const __bf16* __restrict__ Bpk,
                                                       __bf16* __restrict__ out,
                                                       int vt_layout) {
  int gid  = blockIdx.x * blockDim.x + threadIdx.x;
  int lane = gid & 31;
  int wid  = gid >> 5;
  int ntb = wid % NTB;
  int rtb = wid / NTB;
  if (rtb >= RTB) return;

  v8f c[MBLK][NBLK];
#pragma unroll
  for (int mi = 0; mi < MBLK; ++mi)
#pragma unroll
    for (int ni = 0; ni < NBLK; ++ni) c[mi][ni] = {};

  for (int kt = 0; kt < KT; ++kt) {
    v16bf a[MBLK], b[NBLK];
#pragma unroll
    for (int mi = 0; mi < MBLK; ++mi)
      a[mi] = *(const v16bf*)(Apk + (((size_t)(MBLK * rtb + mi) * KT + kt) * 32 + lane) * 16);
#pragma unroll
    for (int ni = 0; ni < NBLK; ++ni)
      b[ni] = *(const v16bf*)(Bpk + (((size_t)kt * NT + NBLK * ntb + ni) * 32 + lane) * 16);
#pragma unroll
    for (int mi = 0; mi < MBLK; ++mi)
#pragma unroll
      for (int ni = 0; ni < NBLK; ++ni)
        c[mi][ni] = WMMA_BF16(a[mi], b[ni], c[mi][ni]);
  }

  int h = lane >> 4, n = lane & 15;
#pragma unroll
  for (int mi = 0; mi < MBLK; ++mi) {
#pragma unroll
    for (int ni = 0; ni < NBLK; ++ni) {
#pragma unroll
      for (int j = 0; j < 8; ++j) {
        int gr = (MBLK * rtb + mi) * 16 + 8 * h + j;  // token index over B*S
        int gc = (NBLK * ntb + ni) * 16 + n;          // col in D
        int bb = gr / SEQ, s = gr % SEQ;
        int head = gc / HDIM, hd = gc % HDIM;
        size_t o = vt_layout
            ? (((size_t)(bb * NHEAD + head) * HDIM + hd) * SEQ + s)
            : (((size_t)(bb * NHEAD + head) * SEQ + s) * HDIM + hd);
        out[o] = (__bf16)c[mi][ni][j];
      }
    }
  }
}

// ------------- flash attention: one wave per (b*h, 16-query tile) -------------
__global__ __launch_bounds__(256) void attn_kernel(const __bf16* __restrict__ Q,
                                                   const __bf16* __restrict__ K,
                                                   const __bf16* __restrict__ Vt,
                                                   __bf16* __restrict__ Ctx) {
  __shared__ __bf16 plds_all[8 * 16 * 32];   // 16x32 bf16 P tile per wave
  int lane  = threadIdx.x & 31;
  int wavei = threadIdx.x >> 5;
  int gw = (blockIdx.x * blockDim.x + threadIdx.x) >> 5;
  int qt = gw % QTILES;
  int bh = gw / QTILES;
  if (bh >= BATCH * NHEAD) return;

  const __bf16* Qb = Q  + (size_t)bh * SEQ * HDIM;
  const __bf16* Kb = K  + (size_t)bh * SEQ * HDIM;
  const __bf16* Vb = Vt + (size_t)bh * HDIM * SEQ;
  __bf16* lp = plds_all + wavei * (16 * 32);

  int h = lane >> 4, n = lane & 15;
  const float NEG_INF = -__builtin_inff();

  // Q A-fragments for hd 0..31 and 32..63
  int qrow = qt * 16 + (lane & 15);
  const __bf16* Qrow = Qb + (size_t)qrow * HDIM;
  v16bf aq0, aq1;
  ((v8bf*)&aq0)[0] = *(const v8bf*)(Qrow + 16 * h);
  ((v8bf*)&aq0)[1] = *(const v8bf*)(Qrow + 8 + 16 * h);
  ((v8bf*)&aq1)[0] = *(const v8bf*)(Qrow + 32 + 16 * h);
  ((v8bf*)&aq1)[1] = *(const v8bf*)(Qrow + 40 + 16 * h);

  v8f o0 = {}, o1 = {}, o2 = {}, o3 = {};
  float m[8], l[8];
#pragma unroll
  for (int j = 0; j < 8; ++j) { m[j] = NEG_INF; l[j] = 0.0f; }

  int row_j0 = qt * 16 + 8 * h;          // row = row_j0 + j
  int nkb = qt / 2 + 1;                  // causal: 32-key blocks covering keys <= qt*16+15

  for (int kb = 0; kb < nkb; ++kb) {
    int k0 = kb * 32;
    // --- scores S = Q * K^T  (two 16-key column tiles) ---
    v8f s0 = {}, s1 = {};
    {
      const __bf16* kr0 = Kb + (size_t)(k0 + n) * HDIM;
      v16bf b0 = *(const v16bf*)(kr0 + 16 * h);
      v16bf b1 = *(const v16bf*)(kr0 + 32 + 16 * h);
      s0 = WMMA_BF16(aq0, b0, s0);
      s0 = WMMA_BF16(aq1, b1, s0);
      const __bf16* kr1 = Kb + (size_t)(k0 + 16 + n) * HDIM;
      v16bf b2 = *(const v16bf*)(kr1 + 16 * h);
      v16bf b3 = *(const v16bf*)(kr1 + 32 + 16 * h);
      s1 = WMMA_BF16(aq0, b2, s1);
      s1 = WMMA_BF16(aq1, b3, s1);
    }
    // --- scale (1/sqrt(64)) + causal mask (-inf invariant under positive scale) ---
#pragma unroll
    for (int j = 0; j < 8; ++j) {
      float v0 = s0[j] * 0.125f;
      float v1 = s1[j] * 0.125f;
      if (k0 + n      > row_j0 + j) v0 = NEG_INF;
      if (k0 + 16 + n > row_j0 + j) v1 = NEG_INF;
      s0[j] = v0; s1[j] = v1;
    }
    // --- online softmax: row max/sum live across 16 lanes of each half ---
    float alpha[8];
#pragma unroll
    for (int j = 0; j < 8; ++j) {
      float lm = fmaxf(s0[j], s1[j]);
      lm = fmaxf(lm, __shfl_xor(lm, 1, 32));
      lm = fmaxf(lm, __shfl_xor(lm, 2, 32));
      lm = fmaxf(lm, __shfl_xor(lm, 4, 32));
      lm = fmaxf(lm, __shfl_xor(lm, 8, 32));
      float nm = fmaxf(m[j], lm);
      alpha[j] = __expf(m[j] - nm);
      m[j] = nm;
    }
#pragma unroll
    for (int j = 0; j < 8; ++j) {
      float p0 = __expf(s0[j] - m[j]);
      float p1 = __expf(s1[j] - m[j]);
      s0[j] = p0; s1[j] = p1;
      float rs = p0 + p1;
      rs += __shfl_xor(rs, 1, 32);
      rs += __shfl_xor(rs, 2, 32);
      rs += __shfl_xor(rs, 4, 32);
      rs += __shfl_xor(rs, 8, 32);
      l[j] = l[j] * alpha[j] + rs;
      o0[j] *= alpha[j]; o1[j] *= alpha[j]; o2[j] *= alpha[j]; o3[j] *= alpha[j];
    }
    // --- P: C-fragment -> A-fragment via LDS bounce (wave-private region) ---
#pragma unroll
    for (int j = 0; j < 8; ++j) {
      lp[(8 * h + j) * 32 + n]      = (__bf16)s0[j];
      lp[(8 * h + j) * 32 + 16 + n] = (__bf16)s1[j];
    }
    v16bf ap;
    {
      int pr = lane & 15;
      ((v8bf*)&ap)[0] = *(const v8bf*)(lp + pr * 32 + 16 * h);
      ((v8bf*)&ap)[1] = *(const v8bf*)(lp + pr * 32 + 8 + 16 * h);
    }
    // --- O += P * V (4 hd tiles of 16) ---
    {
      v16bf bv0 = *(const v16bf*)(Vb + (size_t)(0 * 16 + n) * SEQ + k0 + 16 * h);
      o0 = WMMA_BF16(ap, bv0, o0);
      v16bf bv1 = *(const v16bf*)(Vb + (size_t)(1 * 16 + n) * SEQ + k0 + 16 * h);
      o1 = WMMA_BF16(ap, bv1, o1);
      v16bf bv2 = *(const v16bf*)(Vb + (size_t)(2 * 16 + n) * SEQ + k0 + 16 * h);
      o2 = WMMA_BF16(ap, bv2, o2);
      v16bf bv3 = *(const v16bf*)(Vb + (size_t)(3 * 16 + n) * SEQ + k0 + 16 * h);
      o3 = WMMA_BF16(ap, bv3, o3);
    }
  }

  // normalize and write ctx row-major [B*S, D] as bf16
#pragma unroll
  for (int j = 0; j < 8; ++j) {
    float inv = 1.0f / l[j];
    o0[j] *= inv; o1[j] *= inv; o2[j] *= inv; o3[j] *= inv;
  }
  int bb = bh / NHEAD, head = bh % NHEAD;
#pragma unroll
  for (int j = 0; j < 8; ++j) {
    int tok = qt * 16 + 8 * h + j;
    size_t base = ((size_t)bb * SEQ + tok) * DIM + head * HDIM + n;
    Ctx[base +  0] = (__bf16)o0[j];
    Ctx[base + 16] = (__bf16)o1[j];
    Ctx[base + 32] = (__bf16)o2[j];
    Ctx[base + 48] = (__bf16)o3[j];
  }
}

// ------------- output GEMM (2x4 register-blocked): ctx_pk x Wo_pk + bo -> f32 out -------------
__global__ __launch_bounds__(256) void gemm_out_kernel(const __bf16* __restrict__ Apk,
                                                       const __bf16* __restrict__ Bpk,
                                                       const float* __restrict__ bias,
                                                       float* __restrict__ out) {
  int gid  = blockIdx.x * blockDim.x + threadIdx.x;
  int lane = gid & 31;
  int wid  = gid >> 5;
  int ntb = wid % NTB;
  int rtb = wid / NTB;
  if (rtb >= RTB) return;

  v8f c[MBLK][NBLK];
#pragma unroll
  for (int mi = 0; mi < MBLK; ++mi)
#pragma unroll
    for (int ni = 0; ni < NBLK; ++ni) c[mi][ni] = {};

  for (int kt = 0; kt < KT; ++kt) {
    v16bf a[MBLK], b[NBLK];
#pragma unroll
    for (int mi = 0; mi < MBLK; ++mi)
      a[mi] = *(const v16bf*)(Apk + (((size_t)(MBLK * rtb + mi) * KT + kt) * 32 + lane) * 16);
#pragma unroll
    for (int ni = 0; ni < NBLK; ++ni)
      b[ni] = *(const v16bf*)(Bpk + (((size_t)kt * NT + NBLK * ntb + ni) * 32 + lane) * 16);
#pragma unroll
    for (int mi = 0; mi < MBLK; ++mi)
#pragma unroll
      for (int ni = 0; ni < NBLK; ++ni)
        c[mi][ni] = WMMA_BF16(a[mi], b[ni], c[mi][ni]);
  }

  int h = lane >> 4, n = lane & 15;
#pragma unroll
  for (int mi = 0; mi < MBLK; ++mi) {
#pragma unroll
    for (int ni = 0; ni < NBLK; ++ni) {
#pragma unroll
      for (int j = 0; j < 8; ++j) {
        int gr = (MBLK * rtb + mi) * 16 + 8 * h + j;
        int gc = (NBLK * ntb + ni) * 16 + n;
        out[(size_t)gr * DIM + gc] = c[mi][ni][j] + bias[gc];
      }
    }
  }
}

extern "C" void kernel_launch(void* const* d_in, const int* in_sizes, int n_in,
                              void* d_out, int out_size, void* d_ws, size_t ws_size,
                              hipStream_t stream) {
  (void)in_sizes; (void)n_in; (void)out_size; (void)ws_size;
  const float* X  = (const float*)d_in[0];
  const float* Wq = (const float*)d_in[1];
  const float* Wk = (const float*)d_in[2];
  const float* Wv = (const float*)d_in[3];
  const float* Wo = (const float*)d_in[4];
  const float* bo = (const float*)d_in[5];
  float* out = (float*)d_out;

  char* ws = (char*)d_ws;
  const size_t WPSZ  = (size_t)DIM * DIM * 2;     // 2 MB per packed weight
  const size_t ACTSZ = (size_t)ROWS * DIM * 2;    // 16 MB per bf16 activation
  __bf16* Wqp = (__bf16*)(ws + 0 * WPSZ);
  __bf16* Wkp = (__bf16*)(ws + 1 * WPSZ);
  __bf16* Wvp = (__bf16*)(ws + 2 * WPSZ);
  __bf16* Wop = (__bf16*)(ws + 3 * WPSZ);
  __bf16* Xpk = (__bf16*)(ws + 4 * WPSZ);
  __bf16* Qd  = (__bf16*)(ws + 4 * WPSZ + 1 * ACTSZ);
  __bf16* Kd  = (__bf16*)(ws + 4 * WPSZ + 2 * ACTSZ);
  __bf16* Vtd = (__bf16*)(ws + 4 * WPSZ + 3 * ACTSZ);
  __bf16* Ctx = (__bf16*)(ws + 4 * WPSZ + 4 * ACTSZ);
  __bf16* Cpk = Xpk;  // Xpk dead after projections -> reuse for packed ctx

  dim3 blk(256);
  // pack weights: (DIM/32)*(DIM/16)*32 = 65536 threads
  pack_b_kernel<<<256, blk, 0, stream>>>(Wq, Wqp, DIM, DIM);
  pack_b_kernel<<<256, blk, 0, stream>>>(Wk, Wkp, DIM, DIM);
  pack_b_kernel<<<256, blk, 0, stream>>>(Wv, Wvp, DIM, DIM);
  pack_b_kernel<<<256, blk, 0, stream>>>(Wo, Wop, DIM, DIM);
  // pack X: (ROWS/16)*(DIM/32)*32 = 524288 threads
  pack_a_kernel<float><<<2048, blk, 0, stream>>>(X, Xpk, ROWS, DIM);
  // projections: RTB*NTB = 4096 waves -> 512 blocks of 8 waves
  gemm_qkv_kernel<<<512, blk, 0, stream>>>(Xpk, Wqp, Qd, 0);
  gemm_qkv_kernel<<<512, blk, 0, stream>>>(Xpk, Wkp, Kd, 0);
  gemm_qkv_kernel<<<512, blk, 0, stream>>>(Xpk, Wvp, Vtd, 1);
  // attention: B*H*(S/16) = 8192 waves -> 1024 blocks
  attn_kernel<<<1024, blk, 0, stream>>>(Qd, Kd, Vtd, Ctx);
  // pack ctx and final projection + bias
  pack_a_kernel<__bf16><<<2048, blk, 0, stream>>>(Ctx, Cpk, ROWS, DIM);
  gemm_out_kernel<<<512, blk, 0, stream>>>(Cpk, Wop, bo, out);
}
